// MultiheadAttention_67662914781604
// MI455X (gfx1250) — matrix operand order repaired and checked by reading, not compiled, not run on hardware
//
#include <hip/hip_runtime.h>

#define D_MODEL 1024
#define SEQ     2048
#define BATCH   4
#define NHEAD   16
#define HDIM    64
#define MROWS   8192  // BATCH*SEQ

typedef __attribute__((ext_vector_type(16))) _Float16 v16h;
typedef __attribute__((ext_vector_type(8)))  float    v8f;
typedef __attribute__((ext_vector_type(4)))  _Float16 v4h;

union FragH {
  v16h v;
  uint4 q[2];
};

__device__ __forceinline__ v8f wmma_f16(const FragH& a, const FragH& b, v8f c) {
  // D = A(16x32 f16) x B(32x16 f16) + C(16x16 f32)
  return __builtin_amdgcn_wmma_f32_16x16x32_f16(false, a.v, false, b.v,
                                                (short)0, c, false, false);
}

// ---- CDNA5 async memory->LDS copy (ASYNCcnt-tracked, no VGPR staging) ----
__device__ __forceinline__ void async_load16(const _Float16* lds_dst,
                                             const _Float16* gsrc) {
  // Generic pointers to LDS carry the wave-relative LDS offset in addr[31:0].
  unsigned int loff = (unsigned int)(unsigned long long)lds_dst;
  unsigned long long ga = (unsigned long long)gsrc;
  asm volatile("global_load_async_to_lds_b128 %0, %1, off"
               :
               : "v"(loff), "v"(ga)
               : "memory");
}
__device__ __forceinline__ void wait_async0() {
  asm volatile("s_wait_asynccnt 0" ::: "memory");
}

// ---------------------------------------------------------------- convert
__global__ __launch_bounds__(256) void cvt_f32_f16(const float* __restrict__ s,
                                                   _Float16* __restrict__ d,
                                                   int n) {
  int i = (blockIdx.x * 256 + threadIdx.x) * 4;
  if (i >= n) return;
  float4 f = *(const float4*)(s + i);
  v4h o = {(_Float16)f.x, (_Float16)f.y, (_Float16)f.z, (_Float16)f.w};
  *(v4h*)(d + i) = o;
}

// ---------------------------------------------------------------- GEMM
// C[m][n] = sum_k A[m][k] * W[n][k] + bias[n]
// Block tile 128x128, 8 waves as 4(M) x 2(N), wave tile 32x64 (8 accums).
// Double-buffered LDS tiles filled with async loads; k-loop kept rolled so
// the accumulators stay in fixed register tuples (no copy->hazard-NOP chains).
// mode 0: out32[m][n] f32 row-major (final projection)
// mode 1: out16 at [B,H,S,Dh]  (Q / K head layout)
// mode 2: out16 at [B,H,Dh,S]  (V transposed head layout)
#define ATILE (128 * 40)  // elements per buffer (128 rows x 32 k, stride 40)

__global__ __launch_bounds__(256) void gemm_f16_wmma(
    const _Float16* __restrict__ A, const _Float16* __restrict__ Bw,
    const float* __restrict__ bias, _Float16* __restrict__ out16,
    float* __restrict__ out32, int mode) {
  __shared__ __align__(16) _Float16 As[2 * ATILE];
  __shared__ __align__(16) _Float16 Bs[2 * ATILE];
  const int tid = threadIdx.x;
  const int lane = tid & 31;
  const int wv = tid >> 5;  // 8 waves: 4 (M) x 2 (N)
  const int hf = lane >> 4;
  const int l16 = lane & 15;
  const int wm = wv & 3;
  const int wn = wv >> 2;
  const int m0 = blockIdx.y * 128;
  const int n0 = blockIdx.x * 128;

  // loader: each thread moves 2 A rows + 2 B rows worth of 16B chunks
  const int lrow = tid >> 2;     // 0..63
  const int lc = (tid & 3) * 8;  // f16 col: 0,8,16,24

  const v8f vzero = {};
  v8f acc[2][4];
#pragma unroll
  for (int i = 0; i < 2; ++i)
#pragma unroll
    for (int j = 0; j < 4; ++j) acc[i][j] = vzero;

  auto issue_tile = [&](int kc, int buf) {
    const _Float16* ag = &A[(size_t)(m0 + lrow) * D_MODEL + kc + lc];
    const _Float16* bg = &Bw[(size_t)(n0 + lrow) * D_MODEL + kc + lc];
    _Float16* al = &As[buf + lrow * 40 + lc];
    _Float16* bl = &Bs[buf + lrow * 40 + lc];
    async_load16(al, ag);
    async_load16(al + 64 * 40, ag + (size_t)64 * D_MODEL);
    async_load16(bl, bg);
    async_load16(bl + 64 * 40, bg + (size_t)64 * D_MODEL);
  };

  issue_tile(0, 0);

  int bsel = 0;  // element offset of current buffer
#pragma clang loop unroll(disable)
  for (int it = 0; it < D_MODEL / 32; ++it) {
    wait_async0();
    __syncthreads();
    if (it + 1 < D_MODEL / 32) issue_tile((it + 1) * 32, bsel ^ ATILE);

    const _Float16* as = &As[bsel];
    const _Float16* bs = &Bs[bsel];
    bsel ^= ATILE;
    FragH af[2], bf[4];
#pragma unroll
    for (int mt = 0; mt < 2; ++mt) {  // A frag: k = g*16 + hf*8 + j
      int r = wm * 32 + mt * 16 + l16;
      af[mt].q[0] = *(const uint4*)&as[r * 40 + hf * 8];
      af[mt].q[1] = *(const uint4*)&as[r * 40 + 16 + hf * 8];
    }
#pragma unroll
    for (int nt = 0; nt < 4; ++nt) {  // B frag: k = g*16 + hf*8 + j, n = l16
      int r = wn * 64 + nt * 16 + l16;
      bf[nt].q[0] = *(const uint4*)&bs[r * 40 + hf * 8];
      bf[nt].q[1] = *(const uint4*)&bs[r * 40 + 16 + hf * 8];
    }
#pragma unroll
    for (int mt = 0; mt < 2; ++mt)
#pragma unroll
      for (int nt = 0; nt < 4; ++nt)
        acc[mt][nt] = wmma_f16(af[mt], bf[nt], acc[mt][nt]);
  }

#pragma unroll
  for (int mt = 0; mt < 2; ++mt)
#pragma unroll
    for (int nt = 0; nt < 4; ++nt) {
      int n = n0 + wn * 64 + nt * 16 + l16;
      float bval = bias[n];
#pragma unroll
      for (int r = 0; r < 8; ++r) {
        int m = m0 + wm * 32 + mt * 16 + r + 8 * hf;
        float vv = acc[mt][nt][r] + bval;
        if (mode == 0) {
          out32[(size_t)m * D_MODEL + n] = vv;
        } else {
          int bb = m >> 11, s = m & (SEQ - 1);
          int hh = n >> 6, dh = n & (HDIM - 1);
          size_t idx;
          if (mode == 1)
            idx = (((size_t)(bb * NHEAD + hh)) * SEQ + s) * HDIM + dh;
          else
            idx = (((size_t)(bb * NHEAD + hh)) * HDIM + dh) * SEQ + s;
          out16[idx] = (_Float16)vv;
        }
      }
    }
}

// ---------------------------------------------------------------- flash attention
// Qh, Kh: [B,H,S,64] f16;  Vt: [B,H,64,S] f16;  Ctx: [B,S,1024] f16
#define KTILE (64 * 72)  // elements per K or V buffer

__global__ __launch_bounds__(256) void flash_attn_wmma(
    const _Float16* __restrict__ Qh, const _Float16* __restrict__ Kh,
    const _Float16* __restrict__ Vt, _Float16* __restrict__ Ctx) {
  __shared__ __align__(16) _Float16 Ks[2 * KTILE];    // [key][dh]
  __shared__ __align__(16) _Float16 Vs[2 * KTILE];    // [dh][key]
  __shared__ __align__(16) _Float16 Pb[8 * 16 * 72];  // per-wave P relayout
  const int tid = threadIdx.x;
  const int lane = tid & 31;
  const int wv = tid >> 5;
  const int hf = lane >> 4;
  const int l16 = lane & 15;
  const int bh = blockIdx.y;    // 0..63
  const int qblk = blockIdx.x;  // 0..15
  const int b = bh >> 4, h = bh & 15;
  const size_t qkBase = (size_t)bh * SEQ * HDIM;
  const size_t vtBase = (size_t)bh * HDIM * SEQ;
  const int q0 = qblk * 128 + wv * 16;

  // Q fragments straight from global (rows are 128B, loads 16B-aligned)
  FragH qf[2];
  {
    const _Float16* qp = Qh + qkBase + (size_t)(q0 + l16) * HDIM;
    qf[0].q[0] = *(const uint4*)(qp + hf * 8);
    qf[0].q[1] = *(const uint4*)(qp + 16 + hf * 8);
    qf[1].q[0] = *(const uint4*)(qp + 32 + hf * 8);
    qf[1].q[1] = *(const uint4*)(qp + 48 + hf * 8);
  }

  auto issue_tile = [&](int j, int buf) {
#pragma unroll
    for (int p = 0; p < 2; ++p) {
      int u = tid + p * 256;
      int row = u >> 3, c = (u & 7) * 8;
      async_load16(&Ks[buf + row * 72 + c],
                   &Kh[qkBase + (size_t)(j * 64 + row) * HDIM + c]);
      async_load16(&Vs[buf + row * 72 + c],
                   &Vt[vtBase + (size_t)row * SEQ + j * 64 + c]);
    }
  };

  const v8f vzero = {};
  float mrow[8], lrow[8];
  v8f oacc[4];
#pragma unroll
  for (int r = 0; r < 8; ++r) { mrow[r] = -3.0e38f; lrow[r] = 0.f; }
#pragma unroll
  for (int d = 0; d < 4; ++d) oacc[d] = vzero;

  const float scale = 0.125f;  // 1/sqrt(64)

  issue_tile(0, 0);

  int bsel = 0;
#pragma clang loop unroll(disable)
  for (int j = 0; j < SEQ / 64; ++j) {
    wait_async0();
    __syncthreads();
    if (j + 1 < SEQ / 64) issue_tile(j + 1, bsel ^ KTILE);

    const _Float16* ks = &Ks[bsel];
    const _Float16* vs = &Vs[bsel];
    bsel ^= KTILE;

    // scores: S(16 x 64) = Q(16x64) @ K_blk^T
    v8f sa[4];
#pragma unroll
    for (int nb = 0; nb < 4; ++nb) {
      sa[nb] = vzero;
      FragH kf;
      int n = nb * 16 + l16;  // key index within block
      kf.q[0] = *(const uint4*)&ks[n * 72 + hf * 8];
      kf.q[1] = *(const uint4*)&ks[n * 72 + 16 + hf * 8];
      sa[nb] = wmma_f16(qf[0], kf, sa[nb]);
      kf.q[0] = *(const uint4*)&ks[n * 72 + 32 + hf * 8];
      kf.q[1] = *(const uint4*)&ks[n * 72 + 48 + hf * 8];
      sa[nb] = wmma_f16(qf[1], kf, sa[nb]);
    }

    // online softmax (rows live across 16 lanes of each half-wave)
    float mnew[8], corr[8], rsum[8];
#pragma unroll
    for (int r = 0; r < 8; ++r) {
      float mx = fmaxf(fmaxf(sa[0][r], sa[1][r]), fmaxf(sa[2][r], sa[3][r]));
#pragma unroll
      for (int msk = 1; msk < 16; msk <<= 1)
        mx = fmaxf(mx, __shfl_xor(mx, msk, 32));
      mnew[r] = fmaxf(mrow[r], mx);
      corr[r] = __expf((mrow[r] - mnew[r]) * scale);
      mrow[r] = mnew[r];
      rsum[r] = 0.f;
    }
#pragma unroll
    for (int nb = 0; nb < 4; ++nb)
#pragma unroll
      for (int r = 0; r < 8; ++r) {
        float pp = __expf((sa[nb][r] - mnew[r]) * scale);
        Pb[wv * (16 * 72) + (r + 8 * hf) * 72 + nb * 16 + l16] = (_Float16)pp;
        rsum[r] += pp;
      }
#pragma unroll
    for (int r = 0; r < 8; ++r) {
#pragma unroll
      for (int msk = 1; msk < 16; msk <<= 1)
        rsum[r] += __shfl_xor(rsum[r], msk, 32);
      lrow[r] = lrow[r] * corr[r] + rsum[r];
    }
#pragma unroll
    for (int d = 0; d < 4; ++d)
#pragma unroll
      for (int r = 0; r < 8; ++r) oacc[d][r] *= corr[r];

    // wave-private C-layout -> A-layout relayout; DS is in-order per wave
    asm volatile("s_wait_dscnt 0" ::: "memory");

    FragH pf[2];
    const _Float16* pr = &Pb[wv * (16 * 72) + l16 * 72];
    pf[0].q[0] = *(const uint4*)(pr + hf * 8);
    pf[0].q[1] = *(const uint4*)(pr + 16 + hf * 8);
    pf[1].q[0] = *(const uint4*)(pr + 32 + hf * 8);
    pf[1].q[1] = *(const uint4*)(pr + 48 + hf * 8);

    // O(16 x 64) += P(16x64) @ V_blk(64x64)
#pragma unroll
    for (int d = 0; d < 4; ++d) {
      FragH vf;
      int n = d * 16 + l16;  // dh index
      vf.q[0] = *(const uint4*)&vs[n * 72 + hf * 8];
      vf.q[1] = *(const uint4*)&vs[n * 72 + 16 + hf * 8];
      oacc[d] = wmma_f16(pf[0], vf, oacc[d]);
      vf.q[0] = *(const uint4*)&vs[n * 72 + 32 + hf * 8];
      vf.q[1] = *(const uint4*)&vs[n * 72 + 48 + hf * 8];
      oacc[d] = wmma_f16(pf[1], vf, oacc[d]);
    }
  }

#pragma unroll
  for (int r = 0; r < 8; ++r) {
    float inv = 1.0f / lrow[r];
    int s = q0 + r + 8 * hf;
#pragma unroll
    for (int d = 0; d < 4; ++d) {
      Ctx[((size_t)b * SEQ + s) * D_MODEL + h * HDIM + d * 16 + l16] =
          (_Float16)(oacc[d][r] * inv);
    }
  }
}

// ---------------------------------------------------------------- launch
extern "C" void kernel_launch(void* const* d_in, const int* in_sizes, int n_in,
                              void* d_out, int out_size, void* d_ws,
                              size_t ws_size, hipStream_t stream) {
  (void)in_sizes; (void)n_in; (void)out_size; (void)ws_size;
  const float* q  = (const float*)d_in[0];
  const float* k  = (const float*)d_in[1];
  const float* v  = (const float*)d_in[2];
  const float* Wq = (const float*)d_in[3];
  const float* bq = (const float*)d_in[4];
  const float* Wk = (const float*)d_in[5];
  const float* bk = (const float*)d_in[6];
  const float* Wv = (const float*)d_in[7];
  const float* bv = (const float*)d_in[8];
  const float* Wo = (const float*)d_in[9];
  const float* bo = (const float*)d_in[10];

  _Float16* ws = (_Float16*)d_ws;
  const size_t NX = (size_t)MROWS * D_MODEL;    // 8 Mi elements
  const size_t NW = (size_t)D_MODEL * D_MODEL;  // 1 Mi elements
  _Float16* Xq   = ws;
  _Float16* Xk   = Xq + NX;
  _Float16* Xv   = Xk + NX;
  _Float16* Wq16 = Xv + NX;
  _Float16* Wk16 = Wq16 + NW;
  _Float16* Wv16 = Wk16 + NW;
  _Float16* Wo16 = Wv16 + NW;
  _Float16* Qh   = Wo16 + NW;
  _Float16* Kh   = Qh + NX;
  _Float16* Vt   = Kh + NX;
  _Float16* Ctx  = Vt + NX;  // total ~120 MB of ws

  const int TB = 256;
  int gx = (int)(NX / 4 / TB);
  int gw = (int)(NW / 4 / TB);
  cvt_f32_f16<<<gx, TB, 0, stream>>>(q, Xq, (int)NX);
  cvt_f32_f16<<<gx, TB, 0, stream>>>(k, Xk, (int)NX);
  cvt_f32_f16<<<gx, TB, 0, stream>>>(v, Xv, (int)NX);
  cvt_f32_f16<<<gw, TB, 0, stream>>>(Wq, Wq16, (int)NW);
  cvt_f32_f16<<<gw, TB, 0, stream>>>(Wk, Wk16, (int)NW);
  cvt_f32_f16<<<gw, TB, 0, stream>>>(Wv, Wv16, (int)NW);
  cvt_f32_f16<<<gw, TB, 0, stream>>>(Wo, Wo16, (int)NW);

  dim3 gg(D_MODEL / 128, MROWS / 128);
  gemm_f16_wmma<<<gg, TB, 0, stream>>>(Xq, Wq16, bq, Qh, nullptr, 1);
  gemm_f16_wmma<<<gg, TB, 0, stream>>>(Xk, Wk16, bk, Kh, nullptr, 1);
  gemm_f16_wmma<<<gg, TB, 0, stream>>>(Xv, Wv16, bv, Vt, nullptr, 2);

  flash_attn_wmma<<<dim3(SEQ / 128, BATCH * NHEAD), TB, 0, stream>>>(Qh, Kh, Vt,
                                                                     Ctx);

  gemm_f16_wmma<<<gg, TB, 0, stream>>>(Ctx, Wo16, bo, nullptr, (float*)d_out,
                                       0);
}
